// PointTransformer_16999480557972
// MI455X (gfx1250) — matrix operand order repaired and checked by reading, not compiled, not run on hardware
//
#include <hip/hip_runtime.h>
#include <hip/hip_bf16.h>
#include <math.h>
#include <stdint.h>

// ---------------------------------------------------------------------------
// PointTransformer forward for MI455X (gfx1250).
// All linear layers run through bf16 WMMA GEMMs (v_wmma_f32_16x16x32_bf16,
// fp32 accumulate, fused bias+ReLU).
//  - fast path (M%64==0, N%16==0, K%32==0): 4 waves / 64x16 tile, B-tile
//    staged via global_load_async_to_lds_b128 (double buffered, ASYNCcnt
//    pipelined), A fragments as direct global_load_b128.
//  - generic path: bounds-checked LDS staging for K=3 / tiny-M layers.
// Irregular parts (KNN, FPS, softmax aggregation) are scalar wave32 kernels.
// ---------------------------------------------------------------------------

typedef __bf16 bf16;
typedef __attribute__((ext_vector_type(16))) __bf16 v16bf;
typedef __attribute__((ext_vector_type(8)))  __bf16 v8bf;
typedef __attribute__((ext_vector_type(8)))  float  v8f;

static inline int cdiv(int a, int b) { return (a + b - 1) / b; }

// ---------------- elementwise f32 -> bf16 ----------------
__global__ void f2bf_kernel(const float* __restrict__ src, bf16* __restrict__ dst, int n) {
  int i = blockIdx.x * blockDim.x + threadIdx.x;
  if (i < n) dst[i] = (bf16)src[i];
}

// ---------------------------------------------------------------------------
// Fast WMMA GEMM: Y[M,N] = act(X[M,K] @ W[N,K]^T + bias)
// Requires M%64==0, N%16==0, K%32==0. Block = 4 waves; wave w owns rows
// [blk*64 + w*16, +16), all waves share one 16x32 B tile per K-chunk.
// B tile is brought in with async global->LDS B128 ops, double buffered.
// ---------------------------------------------------------------------------
__global__ __launch_bounds__(128)
void gemm_bf16_wmma_fast(const bf16* __restrict__ X, const bf16* __restrict__ W,
                         const float* __restrict__ bias,
                         float* __restrict__ Yf, bf16* __restrict__ Yb,
                         int M, int N, int K, int relu) {
  __shared__ bf16 Bs[2][16][32];           // double-buffered B tile (2 x 1KB)
  const int tx = threadIdx.x;              // lane 0..31 (wave32)
  const int ty = threadIdx.y;              // wave 0..3
  const int tid = ty * 32 + tx;
  const int row0 = blockIdx.x * 64 + ty * 16;
  const int col0 = blockIdx.y * 16;
  const int ntiles = K >> 5;

  // async B staging: 64 chunks of 16B (16 rows x 4 quads), threads 0..63 issue
  const int brow = tid >> 2;               // 0..15 (valid for tid<64)
  const int bquad = tid & 3;               // 0..3
  const bool loader = tid < 64;
  const bf16* wrow = W + (size_t)(col0 + brow) * K + (bquad << 3);

  const int h = tx >> 4;                   // half-wave select
  const int m = tx & 15;                   // A row / B col within tile
  const bf16* xrow = X + (size_t)(row0 + m) * K;

  v8f acc = {};

  // issue tile t into buffer t&1
  auto issue = [&](int t) {
    if (loader) {
      const bf16* gp = wrow + (t << 5);
      unsigned lds = (unsigned)(uintptr_t)&Bs[t & 1][brow][bquad << 3];
      asm volatile("global_load_async_to_lds_b128 %0, %1, off"
                   :: "v"(lds), "v"((unsigned long long)(uintptr_t)gp)
                   : "memory");
    }
  };

  issue(0);
  for (int t = 0; t < ntiles; ++t) {
    if (t + 1 < ntiles) {
      issue(t + 1);                                        // keep next tile in flight
      asm volatile("s_wait_asynccnt 0x1" ::: "memory");    // wait tile t only
    } else {
      asm volatile("s_wait_asynccnt 0x0" ::: "memory");
    }
    __syncthreads();                                       // publish buffer t&1
    const int k0 = t << 5;
    // A fragment: one contiguous 16-half run per lane (K = 16h..16h+15, row = m)
    v8bf alo = *(const v8bf*)(xrow + k0 + 16 * h);
    v8bf ahi = *(const v8bf*)(xrow + k0 + 16 * h + 8);
    v16bf a = __builtin_shufflevector(alo, ahi, 0, 1, 2, 3, 4, 5, 6, 7,
                                      8, 9, 10, 11, 12, 13, 14, 15);
    // B fragment: one contiguous 32B run per lane (K = 16h..16h+15, col = m)
    v16bf b = *(const v16bf*)&Bs[t & 1][m][16 * h];
    acc = __builtin_amdgcn_wmma_f32_16x16x32_bf16(
        /*neg_a=*/false, a, /*neg_b=*/false, b,
        /*c_mod=*/(short)0, acc, /*reuse_a=*/false, /*reuse_b=*/false);
    __syncthreads();                                       // done reading before restage
  }

  const int col = col0 + m;
  const float bv = bias ? bias[col] : 0.0f;
#pragma unroll
  for (int r = 0; r < 8; ++r) {
    const int row = row0 + r + 8 * h;
    float v = acc[r] + bv;
    if (relu) v = fmaxf(v, 0.0f);
    const size_t o = (size_t)row * N + col;
    if (Yf) Yf[o] = v;
    if (Yb) Yb[o] = (bf16)v;
  }
}

// ---------------------------------------------------------------------------
// Generic WMMA GEMM (bounds-checked; used for K=3 and tiny-M layers).
// Clamp-and-select staging (no per-element EXEC branching).
// ---------------------------------------------------------------------------
__global__ __launch_bounds__(32)
void gemm_bf16_wmma(const bf16* __restrict__ X, const bf16* __restrict__ W,
                    const float* __restrict__ bias,
                    float* __restrict__ Yf, bf16* __restrict__ Yb,
                    int M, int N, int K, int relu) {
  __shared__ bf16 As[16][32];
  __shared__ bf16 Bs[16][32];
  const int lane = threadIdx.x;
  const int row0 = blockIdx.x * 16;
  const int col0 = blockIdx.y * 16;
  v8f acc = {};
  for (int k0 = 0; k0 < K; k0 += 32) {
    const int gk = k0 + lane;
    const int gkc = gk < K ? gk : K - 1;
    const bool kok = gk < K;
#pragma unroll
    for (int t = 0; t < 16; ++t) {
      const int ga = row0 + t, gac = ga < M ? ga : M - 1;
      const int gb = col0 + t, gbc = gb < N ? gb : N - 1;
      const bf16 av = X[(size_t)gac * K + gkc];
      const bf16 bv = W[(size_t)gbc * K + gkc];
      As[t][lane] = (kok && ga < M) ? av : (bf16)0.0f;
      Bs[t][lane] = (kok && gb < N) ? bv : (bf16)0.0f;
    }
    __syncthreads();
    const int h = lane >> 4;
    const int m = lane & 15;
    v16bf a = {}, b = {};
#pragma unroll
    for (int v = 0; v < 8; ++v) {
      const int ka = (v < 4) ? (8 * h + 2 * v) : (16 + 8 * h + 2 * (v - 4));
      a[2 * v]     = As[m][ka];
      a[2 * v + 1] = As[m][ka + 1];
      const int kb = (v < 4) ? (8 * h + 2 * v) : (16 + 8 * h + 2 * (v - 4));
      b[2 * v]     = Bs[m][kb];
      b[2 * v + 1] = Bs[m][kb + 1];
    }
    acc = __builtin_amdgcn_wmma_f32_16x16x32_bf16(
        false, a, false, b, (short)0, acc, false, false);
    __syncthreads();
  }
  const int h = lane >> 4;
  const int col = col0 + (lane & 15);
  const float bv = (bias != nullptr && col < N) ? bias[col] : 0.0f;
#pragma unroll
  for (int r = 0; r < 8; ++r) {
    const int row = row0 + r + 8 * h;
    if (row < M && col < N) {
      float v = acc[r] + bv;
      if (relu) v = fmaxf(v, 0.0f);
      const size_t o = (size_t)row * N + col;
      if (Yf) Yf[o] = v;
      if (Yb) Yb[o] = (bf16)v;
    }
  }
}

// ---------------- KNN (self graph, k=16 + self loop appended) --------------
__global__ void knn_self_kernel(const float* __restrict__ pos, int* __restrict__ nbr, int N) {
  const int b = blockIdx.y;
  const int i = blockIdx.x * blockDim.x + threadIdx.x;
  __shared__ float sp[256][3];
  const float* pb = pos + (size_t)b * N * 3;
  float qx = 0.f, qy = 0.f, qz = 0.f;
  if (i < N) { qx = pb[i * 3]; qy = pb[i * 3 + 1]; qz = pb[i * 3 + 2]; }
  float bd[16]; int bi[16];
#pragma unroll
  for (int t = 0; t < 16; ++t) { bd[t] = 3.0e38f; bi[t] = 0; }
  for (int j0 = 0; j0 < N; j0 += 256) {
    const int j = j0 + threadIdx.x;
    if (j < N) {
      sp[threadIdx.x][0] = pb[j * 3];
      sp[threadIdx.x][1] = pb[j * 3 + 1];
      sp[threadIdx.x][2] = pb[j * 3 + 2];
    }
    __syncthreads();
    const int cnt = (N - j0 < 256) ? (N - j0) : 256;
    if (i < N) {
      for (int jj = 0; jj < cnt; ++jj) {
        const int jg = j0 + jj;
        if (jg == i) continue;  // loop=False
        const float dx = qx - sp[jj][0], dy = qy - sp[jj][1], dz = qz - sp[jj][2];
        const float d = dx * dx + dy * dy + dz * dz;
        if (d < bd[15]) {
          int p = 15;
          while (p > 0 && bd[p - 1] > d) { bd[p] = bd[p - 1]; bi[p] = bi[p - 1]; --p; }
          bd[p] = d; bi[p] = jg;
        }
      }
    }
    __syncthreads();
  }
  if (i < N) {
    int* o = nbr + ((size_t)b * N + i) * 17;
    for (int t = 0; t < 16; ++t) o[t] = bi[t];
    o[16] = i;  // self loop
  }
}

// ---------------- KNN (cross: centers vs full cloud, k=16) -----------------
__global__ void knn_cross_kernel(const float* __restrict__ pos_q, const float* __restrict__ pos,
                                 int* __restrict__ nbr, int Mq, int N) {
  const int b = blockIdx.y;
  const int i = blockIdx.x * blockDim.x + threadIdx.x;
  __shared__ float sp[256][3];
  const float* pb = pos + (size_t)b * N * 3;
  float qx = 0.f, qy = 0.f, qz = 0.f;
  if (i < Mq) {
    const float* q = pos_q + ((size_t)b * Mq + i) * 3;
    qx = q[0]; qy = q[1]; qz = q[2];
  }
  float bd[16]; int bi[16];
#pragma unroll
  for (int t = 0; t < 16; ++t) { bd[t] = 3.0e38f; bi[t] = 0; }
  for (int j0 = 0; j0 < N; j0 += 256) {
    const int j = j0 + threadIdx.x;
    if (j < N) {
      sp[threadIdx.x][0] = pb[j * 3];
      sp[threadIdx.x][1] = pb[j * 3 + 1];
      sp[threadIdx.x][2] = pb[j * 3 + 2];
    }
    __syncthreads();
    const int cnt = (N - j0 < 256) ? (N - j0) : 256;
    if (i < Mq) {
      for (int jj = 0; jj < cnt; ++jj) {
        const float dx = qx - sp[jj][0], dy = qy - sp[jj][1], dz = qz - sp[jj][2];
        const float d = dx * dx + dy * dy + dz * dz;
        if (d < bd[15]) {
          int p = 15;
          while (p > 0 && bd[p - 1] > d) { bd[p] = bd[p - 1]; bi[p] = bi[p - 1]; --p; }
          bd[p] = d; bi[p] = j0 + jj;
        }
      }
    }
    __syncthreads();
  }
  if (i < Mq) {
    int* o = nbr + ((size_t)b * Mq + i) * 16;
    for (int t = 0; t < 16; ++t) o[t] = bi[t];
  }
}

// ---------------- farthest point sampling (one workgroup per cloud) --------
__global__ void fps_kernel(const float* __restrict__ pos, int* __restrict__ idc, int N, int m) {
  const int b = blockIdx.x;
  const int tid = threadIdx.x;
  __shared__ float mind[4096];
  __shared__ float rv[256];
  __shared__ int ri[256];
  __shared__ int s_last;
  const float* pb = pos + (size_t)b * N * 3;
  for (int i = tid; i < N; i += 256) mind[i] = 3.0e38f;
  if (tid == 0) s_last = 0;
  __syncthreads();
  for (int s = 0; s < m; ++s) {
    const int last = s_last;
    if (tid == 0) idc[(size_t)b * m + s] = last;
    const float lx = pb[last * 3], ly = pb[last * 3 + 1], lz = pb[last * 3 + 2];
    float bv = -1.0f; int bidx = 0;
    for (int i = tid; i < N; i += 256) {
      const float dx = pb[i * 3] - lx, dy = pb[i * 3 + 1] - ly, dz = pb[i * 3 + 2] - lz;
      const float d = dx * dx + dy * dy + dz * dz;
      const float nm = fminf(mind[i], d);
      mind[i] = nm;
      if (nm > bv) { bv = nm; bidx = i; }
    }
    rv[tid] = bv; ri[tid] = bidx;
    __syncthreads();
    for (int off = 128; off > 0; off >>= 1) {
      if (tid < off && rv[tid + off] > rv[tid]) { rv[tid] = rv[tid + off]; ri[tid] = ri[tid + off]; }
      __syncthreads();
    }
    if (tid == 0) s_last = ri[0];
    __syncthreads();
  }
}

// ---------------- gather sampled positions ---------------------------------
__global__ void gather_pos_kernel(const float* __restrict__ pos, const int* __restrict__ idc,
                                  float* __restrict__ out, int N, int m, int total) {
  const int id = blockIdx.x * blockDim.x + threadIdx.x;
  if (id >= total) return;
  const int b = id / m;
  const int j = idc[id];
  const float* s = pos + ((size_t)b * N + j) * 3;
  float* d = out + (size_t)id * 3;
  d[0] = s[0]; d[1] = s[1]; d[2] = s[2];
}

// ---------------- neighborhood max-pool (downsample) -----------------------
__global__ void gather_max_kernel(const float* __restrict__ xf, const int* __restrict__ nbr,
                                  bf16* __restrict__ out_b, int N, int m, int C, int total) {
  const int id = blockIdx.x * blockDim.x + threadIdx.x;
  if (id >= total) return;  // total = B*m*C
  const int c = id % C;
  const int r = id / C;
  const int i = r % m;
  const int b = r / m;
  const int* nb = nbr + ((size_t)b * m + i) * 16;
  float mx = -3.0e38f;
  for (int k = 0; k < 16; ++k) {
    const int j = nb[k];
    mx = fmaxf(mx, xf[((size_t)b * N + j) * C + c]);
  }
  out_b[id] = (bf16)mx;
}

// ---------------- per-edge relative positions (bf16, feeds pos_nn) ---------
__global__ void build_relpos_kernel(const float* __restrict__ pos, const int* __restrict__ nbr,
                                    bf16* __restrict__ rel, int N, int Kn, int total) {
  const int e = blockIdx.x * blockDim.x + threadIdx.x;
  if (e >= total) return;  // total = B*N*Kn
  const int r = e / Kn;    // b*N + i
  const int b = r / N;
  const int i = r % N;
  const int j = nbr[e];
  const float* pi = pos + ((size_t)b * N + i) * 3;
  const float* pj = pos + ((size_t)b * N + j) * 3;
  bf16* d = rel + (size_t)e * 3;
  d[0] = (bf16)(pi[0] - pj[0]);
  d[1] = (bf16)(pi[1] - pj[1]);
  d[2] = (bf16)(pi[2] - pj[2]);
}

// ---------------- per-edge attention input: q_i - s_j + delta --------------
__global__ void build_attn_kernel(const float* __restrict__ q, const float* __restrict__ s,
                                  const float* __restrict__ delta, const int* __restrict__ nbr,
                                  bf16* __restrict__ out, int N, int C, int Kn, long long total) {
  const long long id = (long long)blockIdx.x * blockDim.x + threadIdx.x;
  if (id >= total) return;  // total = E*C
  const int c = (int)(id % C);
  const long long e = id / C;
  const int r = (int)(e / Kn);  // b*N + i
  const int b = r / N;
  const int j = nbr[e];
  const float v = q[(size_t)r * C + c] - s[((size_t)b * N + j) * C + c] + delta[(size_t)id];
  out[(size_t)id] = (bf16)v;
}

// ---------------- softmax over neighbors + weighted aggregate --------------
__global__ void softmax_agg_kernel(const float* __restrict__ alpha, const float* __restrict__ v,
                                   const float* __restrict__ delta, const int* __restrict__ nbr,
                                   bf16* __restrict__ out_b, int N, int C, int Kn, int total) {
  const int id = blockIdx.x * blockDim.x + threadIdx.x;
  if (id >= total) return;  // total = B*N*C
  const int c = id % C;
  const int r = id / C;     // b*N + i
  const int b = r / N;
  const size_t eb = (size_t)r * Kn;
  float mx = -3.0e38f;
  for (int k = 0; k < Kn; ++k) mx = fmaxf(mx, alpha[(eb + k) * C + c]);
  float den = 0.0f, num = 0.0f;
  for (int k = 0; k < Kn; ++k) {
    const float w = __expf(alpha[(eb + k) * C + c] - mx);
    const int j = nbr[eb + k];
    num += w * (v[((size_t)b * N + j) * C + c] + delta[(eb + k) * C + c]);
    den += w;
  }
  out_b[id] = (bf16)(num / den);
}

// ---------------- global mean pool -----------------------------------------
__global__ void meanpool_kernel(const float* __restrict__ x, bf16* __restrict__ g,
                                int N, int C, int total) {
  const int id = blockIdx.x * blockDim.x + threadIdx.x;
  if (id >= total) return;  // total = B*C
  const int c = id % C;
  const int b = id / C;
  float s = 0.0f;
  for (int i = 0; i < N; ++i) s += x[((size_t)b * N + i) * C + c];
  g[id] = (bf16)(s / (float)N);
}

// ---------------- log_softmax ----------------------------------------------
__global__ void logsoftmax_kernel(const float* __restrict__ logits, float* __restrict__ out,
                                  int Bn, int C) {
  const int b = threadIdx.x;
  if (b >= Bn) return;
  const float* r = logits + (size_t)b * C;
  float mx = -3.0e38f;
  for (int c = 0; c < C; ++c) mx = fmaxf(mx, r[c]);
  float s = 0.0f;
  for (int c = 0; c < C; ++c) s += __expf(r[c] - mx);
  const float lse = mx + __logf(s);
  for (int c = 0; c < C; ++c) out[(size_t)b * C + c] = r[c] - lse;
}

// ---------------------------------------------------------------------------
// Host-side orchestration
// ---------------------------------------------------------------------------
struct Arena {
  char* base; size_t off;
  void* alloc(size_t bytes) {
    void* p = base + off;
    off = (off + bytes + 255) & ~((size_t)255);
    return p;
  }
};
struct LinP { const float* w; const float* b; };
struct BlockP {
  LinP lin_in, lin_out, pos1, pos2, att1, att2;
  const float *lin_w, *lin_src_w, *lin_dst_w;
};
struct BlockW { bf16 *lin_in, *lin_out, *pos1, *pos2, *att1, *att2, *lin, *lin_src, *lin_dst; };

extern "C" void kernel_launch(void* const* d_in, const int* in_sizes, int n_in,
                              void* d_out, int out_size, void* d_ws, size_t ws_size,
                              hipStream_t stream) {
  (void)in_sizes; (void)n_in; (void)out_size; (void)ws_size;
  const int Bsz = 8, N0 = 4096;
  const int DIMS[5] = {32, 64, 128, 256, 512};

  // ---- parse inputs: x, pos, then params in setup_inputs() insertion order
  int pi = 0;
  auto nf = [&]() -> const float* { return (const float*)d_in[pi++]; };
  const float* x_in = nf();
  const float* pos_in = nf();
  LinP mlp_input{nf(), nf()};
  auto rdBlock = [&]() -> BlockP {
    BlockP p;
    p.lin_in = {nf(), nf()}; p.lin_out = {nf(), nf()};
    p.pos1 = {nf(), nf()};   p.pos2 = {nf(), nf()};
    p.att1 = {nf(), nf()};   p.att2 = {nf(), nf()};
    p.lin_w = nf(); p.lin_src_w = nf(); p.lin_dst_w = nf();
    return p;
  };
  BlockP trin = rdBlock();
  LinP td[4]; for (int i = 0; i < 4; ++i) td[i] = {nf(), nf()};
  BlockP trd[4]; for (int i = 0; i < 4; ++i) trd[i] = rdBlock();
  LinP out1{nf(), nf()}, out2{nf(), nf()}, out3{nf(), nf()};

  Arena ar{(char*)d_ws, 0};

  auto cvt = [&](const float* w, int n) -> bf16* {
    bf16* p = (bf16*)ar.alloc((size_t)n * sizeof(bf16));
    f2bf_kernel<<<cdiv(n, 256), 256, 0, stream>>>(w, p, n);
    return p;
  };
  auto gemm = [&](const bf16* X, const bf16* W, const float* bias, float* Yf, bf16* Yb,
                  int M, int N, int K, int relu) {
    if ((M % 64 == 0) && (N % 16 == 0) && (K % 32 == 0)) {
      dim3 g(M / 64, N / 16);
      gemm_bf16_wmma_fast<<<g, dim3(32, 4), 0, stream>>>(X, W, bias, Yf, Yb, M, N, K, relu);
    } else {
      dim3 g(cdiv(M, 16), cdiv(N, 16));
      gemm_bf16_wmma<<<g, 32, 0, stream>>>(X, W, bias, Yf, Yb, M, N, K, relu);
    }
  };

  // ---- convert all weights to bf16 once (bias stays fp32, fused in GEMM)
  bf16* mlp_w = cvt(mlp_input.w, 32 * 3);
  auto cvtBlock = [&](const BlockP& p, int C) -> BlockW {
    BlockW w;
    w.lin_in = cvt(p.lin_in.w, C * C);   w.lin_out = cvt(p.lin_out.w, C * C);
    w.pos1 = cvt(p.pos1.w, 64 * 3);      w.pos2 = cvt(p.pos2.w, C * 64);
    w.att1 = cvt(p.att1.w, 64 * C);      w.att2 = cvt(p.att2.w, C * 64);
    w.lin = cvt(p.lin_w, C * C);
    w.lin_src = cvt(p.lin_src_w, C * C); w.lin_dst = cvt(p.lin_dst_w, C * C);
    return w;
  };
  BlockW trin_w = cvtBlock(trin, 32);
  bf16* td_w[4];
  for (int i = 0; i < 4; ++i) td_w[i] = cvt(td[i].w, DIMS[i + 1] * DIMS[i]);
  BlockW trd_w[4];
  for (int i = 0; i < 4; ++i) trd_w[i] = cvtBlock(trd[i], DIMS[i + 1]);
  bf16* o1w = cvt(out1.w, 64 * 512);
  bf16* o2w = cvt(out2.w, 64 * 64);
  bf16* o3w = cvt(out3.w, 40 * 64);

  // ---- transformer block (all heavy math via WMMA GEMM)
  auto run_block = [&](const bf16* x_b, const float* pos_l, const int* nbr_l, int N, int C,
                       const BlockP& P, const BlockW& W, float** out_f, bf16** out_b) {
    const int Mn = Bsz * N, E = Mn * 17;
    float* y_f = (float*)ar.alloc((size_t)Mn * C * 4);
    bf16* y_b = (bf16*)ar.alloc((size_t)Mn * C * 2);
    const size_t tmark = ar.off;
    bf16* h_b = (bf16*)ar.alloc((size_t)Mn * C * 2);
    float* q_f = (float*)ar.alloc((size_t)Mn * C * 4);
    float* s_f = (float*)ar.alloc((size_t)Mn * C * 4);
    float* v_f = (float*)ar.alloc((size_t)Mn * C * 4);
    float* delta = (float*)ar.alloc((size_t)E * C * 4);

    gemm(x_b, W.lin_in, P.lin_in.b, nullptr, h_b, Mn, C, C, 1);   // relu(lin_in)
    gemm(h_b, W.lin_dst, nullptr, q_f, nullptr, Mn, C, C, 0);     // alpha_i
    gemm(h_b, W.lin_src, nullptr, s_f, nullptr, Mn, C, C, 0);     // alpha_j
    gemm(h_b, W.lin, nullptr, v_f, nullptr, Mn, C, C, 0);         // value

    {  // pos_nn: 3 -> 64 -> C over all edges
      const size_t m2 = ar.off;
      bf16* rel = (bf16*)ar.alloc((size_t)E * 3 * 2);
      bf16* t1 = (bf16*)ar.alloc((size_t)E * 64 * 2);
      build_relpos_kernel<<<cdiv(E, 256), 256, 0, stream>>>(pos_l, nbr_l, rel, N, 17, E);
      gemm(rel, W.pos1, P.pos1.b, nullptr, t1, E, 64, 3, 1);
      gemm(t1, W.pos2, P.pos2.b, delta, nullptr, E, C, 64, 0);
      ar.off = m2;  // reclaim rel/t1 (stream-ordered reuse is safe)
    }
    float* alpha = (float*)ar.alloc((size_t)E * C * 4);
    bf16* agg_b = (bf16*)ar.alloc((size_t)Mn * C * 2);
    {  // attn_nn: C -> 64 -> C over all edges
      const size_t m3 = ar.off;
      bf16* e_b = (bf16*)ar.alloc((size_t)E * C * 2);
      bf16* t2 = (bf16*)ar.alloc((size_t)E * 64 * 2);
      const long long tot = (long long)E * C;
      build_attn_kernel<<<(int)((tot + 255) / 256), 256, 0, stream>>>(q_f, s_f, delta, nbr_l,
                                                                      e_b, N, C, 17, tot);
      gemm(e_b, W.att1, P.att1.b, nullptr, t2, E, 64, C, 1);
      gemm(t2, W.att2, P.att2.b, alpha, nullptr, E, C, 64, 0);
      ar.off = m3;  // reclaim e_b/t2
    }
    softmax_agg_kernel<<<cdiv(Mn * C, 256), 256, 0, stream>>>(alpha, v_f, delta, nbr_l,
                                                              agg_b, N, C, 17, Mn * C);
    gemm(agg_b, W.lin_out, P.lin_out.b, y_f, y_b, Mn, C, C, 1);   // relu(lin_out)
    ar.off = tmark;  // reclaim all block temporaries
    *out_f = y_f; *out_b = y_b;
  };

  // ---- input MLP: relu(x @ W^T + b), K=3 zero-padded in the GEMM
  bf16* xin_b = (bf16*)ar.alloc((size_t)Bsz * N0 * 3 * 2);
  f2bf_kernel<<<cdiv(Bsz * N0 * 3, 256), 256, 0, stream>>>(x_in, xin_b, Bsz * N0 * 3);
  bf16* x_b = (bf16*)ar.alloc((size_t)Bsz * N0 * 32 * 2);
  gemm(xin_b, mlp_w, mlp_input.b, nullptr, x_b, Bsz * N0, 32, 3, 1);

  // ---- level 0
  int* nbr0 = (int*)ar.alloc((size_t)Bsz * N0 * 17 * 4);
  knn_self_kernel<<<dim3(cdiv(N0, 256), Bsz), 256, 0, stream>>>(pos_in, nbr0, N0);
  float* x_f = nullptr;
  run_block(x_b, pos_in, nbr0, N0, 32, trin, trin_w, &x_f, &x_b);

  // ---- 4 downsampling levels
  const float* pos = pos_in;
  int N = N0;
  for (int lv = 0; lv < 4; ++lv) {
    const int m = N / 4;
    const int Cin = DIMS[lv], Cout = DIMS[lv + 1];
    int* idc = (int*)ar.alloc((size_t)Bsz * m * 4);
    fps_kernel<<<Bsz, 256, 0, stream>>>(pos, idc, N, m);
    float* pos_c = (float*)ar.alloc((size_t)Bsz * m * 3 * 4);
    gather_pos_kernel<<<cdiv(Bsz * m, 256), 256, 0, stream>>>(pos, idc, pos_c, N, m, Bsz * m);
    int* nbr_c = (int*)ar.alloc((size_t)Bsz * m * 16 * 4);
    knn_cross_kernel<<<dim3(cdiv(m, 256), Bsz), 256, 0, stream>>>(pos_c, pos, nbr_c, m, N);
    float* xf = (float*)ar.alloc((size_t)Bsz * N * Cout * 4);
    gemm(x_b, td_w[lv], td[lv].b, xf, nullptr, Bsz * N, Cout, Cin, 1);  // relu(lin(x, td))
    bf16* xn_b = (bf16*)ar.alloc((size_t)Bsz * m * Cout * 2);
    gather_max_kernel<<<cdiv(Bsz * m * Cout, 256), 256, 0, stream>>>(xf, nbr_c, xn_b, N, m,
                                                                     Cout, Bsz * m * Cout);
    int* nbr_n = (int*)ar.alloc((size_t)Bsz * m * 17 * 4);
    knn_self_kernel<<<dim3(cdiv(m, 256), Bsz), 256, 0, stream>>>(pos_c, nbr_n, m);
    run_block(xn_b, pos_c, nbr_n, m, Cout, trd[lv], trd_w[lv], &x_f, &x_b);
    pos = pos_c;
    N = m;
  }

  // ---- global mean pool + 3-layer head + log_softmax
  bf16* g_b = (bf16*)ar.alloc((size_t)Bsz * 512 * 2);
  meanpool_kernel<<<cdiv(Bsz * 512, 256), 256, 0, stream>>>(x_f, g_b, N, 512, Bsz * 512);
  bf16* h1_b = (bf16*)ar.alloc((size_t)Bsz * 64 * 2);
  gemm(g_b, o1w, out1.b, nullptr, h1_b, Bsz, 64, 512, 1);
  bf16* h2_b = (bf16*)ar.alloc((size_t)Bsz * 64 * 2);
  gemm(h1_b, o2w, out2.b, nullptr, h2_b, Bsz, 64, 64, 1);
  float* logits = (float*)ar.alloc((size_t)Bsz * 40 * 4);
  gemm(h2_b, o3w, out3.b, logits, nullptr, Bsz, 40, 64, 0);
  logsoftmax_kernel<<<1, 32, 0, stream>>>(logits, (float*)d_out, Bsz, 40);
}
